// QMultiHeadedAttention_30279519437617
// MI455X (gfx1250) — matrix operand order repaired and checked by reading, not compiled, not run on hardware
//
#include <hip/hip_runtime.h>
#include <hip/hip_bf16.h>

// ---------------------------------------------------------------------------
// QMultiHeadedAttention on MI455X (gfx1250): bf16 WMMA pipeline.
//   1) gemm_xWT<f32in,bf16out>:  kk/vv/qq = x @ W.T + b   (qq scaled by 1/8)
//   2) attn_kernel:              per (b,h) flash attention on the raw-view
//                                slab (K:[64x1024], Q/V:[1024x64]) -> ctx bf16
//   3) gemm_xWT<bf16in,f32out>:  out = ctx @ Wo.T + bo    (A-tile via TDM)
// ---------------------------------------------------------------------------

typedef unsigned short bhalf;
typedef __bf16  bf16x16 __attribute__((ext_vector_type(16)));
typedef __bf16  bf16x4  __attribute__((ext_vector_type(4)));
typedef float   f32x4   __attribute__((ext_vector_type(4)));
typedef float   v8f     __attribute__((ext_vector_type(8)));

union Frag   { bf16x16 v; uint4 q[2]; };
union CvtIn  { float4 f4; f32x4 f; };
union CvtOut { bf16x4 h; uint2 u; };

// compiler-lowered RNE converts (v_cvt_pk_bf16_f32 where available)
__device__ __forceinline__ bhalf f32_to_bf16(float f) {
    __bf16 h = (__bf16)f;
    return __builtin_bit_cast(bhalf, h);
}
__device__ __forceinline__ uint2 cvt4_bf16(float4 f) {
    CvtIn  ci; ci.f4 = f;
    CvtOut co; co.h = __builtin_convertvector(ci.f, bf16x4);
    return co.u;
}

#if __has_builtin(__builtin_amdgcn_tensor_load_to_lds)
#define HAVE_TDM 1
#endif
#if __has_builtin(__builtin_amdgcn_s_wait_tensorcnt)
#define HAVE_WAIT_TENSOR 1
#endif

// ---------------- TDM helper (5-arg clang-22 / 6-arg clang-23) -------------
#if defined(HAVE_TDM)
typedef unsigned int u32x4 __attribute__((ext_vector_type(4)));
typedef int          i32x4 __attribute__((ext_vector_type(4)));
typedef int          i32x8 __attribute__((ext_vector_type(8)));
typedef __attribute__((address_space(3))) bhalf lds_bhalf;

// 2D bf16 tile load: tile_d0 elems/row, tile_d1 rows, row stride (elems),
// LDS pad: pad_int_code dwords-interval code / pad_amt_code dwords code.
__device__ __forceinline__ void tdm_load_2d_bf16(
    const bhalf* gsrc, unsigned lds_off,
    int tile_d0, int tile_d1, int tensor_d0, int tensor_d1, long long stride,
    int pad_int_code, int pad_amt_code)
{
    unsigned long long ga = (unsigned long long)(const void*)gsrc;
    u32x4 g0;
    g0[0] = 1u;                                   // count=1 (valid D#)
    g0[1] = lds_off;                              // lds_addr
    g0[2] = (unsigned)(ga & 0xffffffffu);         // global_addr[31:0]
    g0[3] = (unsigned)((ga >> 32) & 0x01ffffffu)  // global_addr[56:32]
          | (2u << 30);                           // type=2 ("image")
    i32x8 g1;
    g1[0] = (1 << 16)                             // data_size = 2 bytes
          | (1 << 20)                             // pad_enable
          | (pad_int_code << 22)                  // pad interval (dwords)
          | (pad_amt_code << 25);                 // pad amount (dwords)
    g1[1] = (tensor_d0 & 0xffff) << 16;           // tensor_dim0[15:0]
    g1[2] = ((tensor_d0 >> 16) & 0xffff)          // tensor_dim0[31:16]
          | ((tensor_d1 & 0xffff) << 16);         // tensor_dim1[15:0]
    g1[3] = ((tensor_d1 >> 16) & 0xffff)
          | ((tile_d0 & 0xffff) << 16);           // tile_dim0
    g1[4] = (tile_d1 & 0xffff);                   // tile_dim1 (tile_dim2=0)
    g1[5] = (int)(stride & 0xffffffff);           // tensor_dim0_stride lo
    g1[6] = (int)((stride >> 32) & 0xffff);       // tensor_dim0_stride hi
    g1[7] = 0;
    i32x4 z4 = {0, 0, 0, 0};
#if __clang_major__ >= 23
    i32x8 z8 = {0, 0, 0, 0, 0, 0, 0, 0};
    __builtin_amdgcn_tensor_load_to_lds(g0, g1, z4, z4, z8, 0);
#else
    __builtin_amdgcn_tensor_load_to_lds(g0, g1, z4, z4, 0);
#endif
}
__device__ __forceinline__ unsigned lds_offset_of(const bhalf* p) {
    return (unsigned)(unsigned long long)(lds_bhalf*)p;
}
#endif

// ===========================================================================
// GEMM: Y[M=8192, N=1024] = X[M,1024] @ W[N=1024, K=1024]^T + bias, * scale
// Block tile 128x128, BK=32.  8 waves: wave grid 4(m) x 2(n), each wave
// computes 32x64 = 2x4 tiles of v_wmma_f32_16x16x32_bf16.
// ===========================================================================
template<int IN_BF16, int OUT_F32>
__global__ __launch_bounds__(256)
void gemm_xWT(const void* __restrict__ Xv,
              const float* __restrict__ Wm,
              const float* __restrict__ bias,
              void* __restrict__ Yv,
              float out_scale)
{
    __shared__ bhalf As[128][40];   // 128 rows x 32 K (+8 pad), 16B-aligned rows
    __shared__ bhalf Bs[128][40];

    const int tid  = threadIdx.x;
    const int lane = tid & 31;
    const int wv   = tid >> 5;         // 0..7
    const int wm   = wv >> 1;          // 0..3
    const int wn   = wv & 1;           // 0..1
    const int lrow = lane & 15;
    const int lhi  = lane >> 4;        // 0/1
    const int m0   = blockIdx.x * 128;
    const int n0   = blockIdx.y * 128;

    const float* Xf = (const float*)Xv;
    const bhalf* Xb = (const bhalf*)Xv;

    v8f acc[2][4];
    const v8f vzero = {0.f,0.f,0.f,0.f,0.f,0.f,0.f,0.f};
#pragma unroll
    for (int mi = 0; mi < 2; ++mi)
#pragma unroll
        for (int ni = 0; ni < 4; ++ni) acc[mi][ni] = vzero;

    const int srow = tid >> 1;           // 0..127
    const int scol = (tid & 1) * 16;     // 0 / 16

    for (int kb = 0; kb < 1024; kb += 32) {
        __syncthreads();
        // ---- stage A ----
        if constexpr (IN_BF16) {
#if defined(HAVE_TDM)
            // Tensor Data Mover: each wave DMAs its 16x32 bf16 slice of the
            // A tile (row stride 1024 elems).  LDS pad 16+4 dwords matches
            // the 80B padded row of As.
            tdm_load_2d_bf16(Xb + (size_t)(m0 + wv * 16) * 1024 + kb,
                             lds_offset_of(&As[wv * 16][0]),
                             /*tile*/32, 16, /*tensor*/1024, 8192,
                             /*stride*/1024, /*pad 16 dw*/3, /*pad 4 dw*/3);
#if defined(HAVE_WAIT_TENSOR)
            __builtin_amdgcn_s_wait_tensorcnt(0);
#endif
#else
            const uint4* s = (const uint4*)(Xb + (size_t)(m0 + srow) * 1024 + kb + scol);
            *(uint4*)&As[srow][scol]     = s[0];
            *(uint4*)&As[srow][scol + 8] = s[1];
#endif
        } else {
            __attribute__((aligned(16))) bhalf tmp[16];
            const float4* s = (const float4*)(Xf + (size_t)(m0 + srow) * 1024 + kb + scol);
#pragma unroll
            for (int i = 0; i < 4; ++i)
                *(uint2*)&tmp[i * 4] = cvt4_bf16(s[i]);
            *(uint4*)&As[srow][scol]     = *(const uint4*)&tmp[0];
            *(uint4*)&As[srow][scol + 8] = *(const uint4*)&tmp[8];
        }

        // ---- stage B: rows of W (W[n][k], k contiguous -> B = W^T) ----
        {
            const float4* ws = (const float4*)(Wm + (size_t)(n0 + srow) * 1024 + kb + scol);
            __attribute__((aligned(16))) bhalf tw[16];
#pragma unroll
            for (int i = 0; i < 4; ++i)
                *(uint2*)&tw[i * 4] = cvt4_bf16(ws[i]);
            *(uint4*)&Bs[srow][scol]     = *(const uint4*)&tw[0];
            *(uint4*)&Bs[srow][scol + 8] = *(const uint4*)&tw[8];

            if (kb + 32 < 1024) {   // -> global_prefetch_b8
                if (IN_BF16)
                    __builtin_prefetch(Xb + (size_t)(m0 + srow) * 1024 + kb + 32 + scol, 0, 0);
                else
                    __builtin_prefetch(Xf + (size_t)(m0 + srow) * 1024 + kb + 32 + scol, 0, 0);
                __builtin_prefetch(Wm + (size_t)(n0 + srow) * 1024 + kb + 32 + scol, 0, 0);
            }
        }
        __syncthreads();

        // ---- compute: 2x4 WMMA tiles per wave, one K-step of 32 ----
#pragma unroll
        for (int mi = 0; mi < 2; ++mi) {
            const bhalf* ab = &As[wm*32 + mi*16 + lrow][lhi * 8];
            Frag a;
            a.q[0] = *(const uint4*)(ab);
            a.q[1] = *(const uint4*)(ab + 16);
#pragma unroll
            for (int ni = 0; ni < 4; ++ni) {
                const bhalf* bb = &Bs[wn*64 + ni*16 + lrow][lhi * 8];
                Frag bf;
                bf.q[0] = *(const uint4*)(bb);
                bf.q[1] = *(const uint4*)(bb + 16);
                acc[mi][ni] = __builtin_amdgcn_wmma_f32_16x16x32_bf16(
                    false, a.v, false, bf.v, (short)0, acc[mi][ni], false, false);
            }
        }
    }

    // ---- epilogue: bias, scale, store (C layout: m=r+8*lhi, n=lrow) ----
    float* Yf = (float*)Yv;
    bhalf* Yb = (bhalf*)Yv;
#pragma unroll
    for (int mi = 0; mi < 2; ++mi) {
#pragma unroll
        for (int ni = 0; ni < 4; ++ni) {
            const int n = n0 + wn*64 + ni*16 + lrow;
            const float bv = bias[n];
#pragma unroll
            for (int r = 0; r < 8; ++r) {
                const int m = m0 + wm*32 + mi*16 + r + 8*lhi;
                const float val = (acc[mi][ni][r] + bv) * out_scale;
                if (OUT_F32) Yf[(size_t)m * 1024 + n] = val;
                else         Yb[(size_t)m * 1024 + n] = f32_to_bf16(val);
            }
        }
    }
}

// ===========================================================================
// Attention: grid = B(8) * H(16) * (1024/128) row blocks = 1024 workgroups.
// Per (b,h): contiguous 64K-elem slab; K=[64x1024], Q=V=[1024x64] row-major.
// Each wave owns 16 Q rows; flash-style online softmax over 16 column blocks
// of 64.  S and O GEMMs via v_wmma_f32_16x16x32_bf16.
// ===========================================================================
__global__ __launch_bounds__(256)
void attn_kernel(const bhalf* __restrict__ kk, const bhalf* __restrict__ vv,
                 const bhalf* __restrict__ qq, bhalf* __restrict__ ctx)
{
    __shared__ bhalf Qs[128][72];     // Q tile rows (64 used cols + 8 pad)
    __shared__ bhalf Ks[64][72];      // K block transposed: Ks[c][j] = K[j][c]
    __shared__ bhalf Vs[64][72];      // V block transposed: Vs[j][c] = V[c][j]
    __shared__ bhalf Ps[8][16][72];   // per-wave P (softmax probs), bf16

    const int tid  = threadIdx.x;
    const int lane = tid & 31;
    const int wv   = tid >> 5;
    const int lrow = lane & 15;
    const int lhi  = lane >> 4;

    const int wblk = blockIdx.x & 7;
    const int h    = (blockIdx.x >> 3) & 15;
    const int b    = blockIdx.x >> 7;

    const size_t slab = ((size_t)(b * 1024 + h * 64)) * 1024;  // 65536 elems
    const int m0 = wblk * 128;

    // ---------------- load Q tile [128 x 64] into LDS ----------------
#if defined(HAVE_TDM)
    // each wave DMAs its 16 rows (tile 64x16 of the [1024x64] Q view);
    // LDS pad 32+4 dwords matches the 144B padded row of Qs.
    tdm_load_2d_bf16(qq + slab + (size_t)(m0 + wv * 16) * 64,
                     lds_offset_of(&Qs[wv * 16][0]),
                     /*tile*/64, 16, /*tensor*/64, 1024,
                     /*stride*/64, /*pad 32 dw*/4, /*pad 4 dw*/3);
#if defined(HAVE_WAIT_TENSOR)
    __builtin_amdgcn_s_wait_tensorcnt(0);
#endif
#else
    {
        const int row  = tid >> 1;
        const int half = (tid & 1) * 32;
        const uint4* src = (const uint4*)(qq + slab + (size_t)(m0 + row) * 64 + half);
        uint4* dst = (uint4*)&Qs[row][half];
        dst[0] = src[0]; dst[1] = src[1]; dst[2] = src[2]; dst[3] = src[3];
    }
#endif

    v8f oacc[4];
    const v8f vzero = {0.f,0.f,0.f,0.f,0.f,0.f,0.f,0.f};
#pragma unroll
    for (int nt = 0; nt < 4; ++nt) oacc[nt] = vzero;
    float rmax[8], rsum[8];
#pragma unroll
    for (int r = 0; r < 8; ++r) { rmax[r] = -1e30f; rsum[r] = 0.f; }

    // staging coords: row-pair = 2*(tid&31), 8-col segment = (tid>>5)*8.
    // Within a wave all 32 lanes write distinct LDS banks (b32 pair stores).
    const int jp = (tid & 31) * 2;     // row pair 0,2,..,62
    const int sc = (tid >> 5) * 8;     // col segment 0,8,..,56

    for (int cb = 0; cb < 16; ++cb) {
        __syncthreads();
        // ---- stage K block [64j x 64c] -> Ks[c][j]; V block -> Vs[j][c] ----
        {
            const uint4 k0 = *(const uint4*)(kk + slab + (size_t)jp * 1024 + cb * 64 + sc);
            const uint4 k1 = *(const uint4*)(kk + slab + (size_t)(jp + 1) * 1024 + cb * 64 + sc);
            const bhalf* e0 = (const bhalf*)&k0;
            const bhalf* e1 = (const bhalf*)&k1;
#pragma unroll
            for (int t = 0; t < 8; ++t)
                *(unsigned int*)&Ks[sc + t][jp] =
                    (unsigned int)e0[t] | ((unsigned int)e1[t] << 16);

            const uint4 v0 = *(const uint4*)(vv + slab + (size_t)(cb * 64 + jp) * 64 + sc);
            const uint4 v1 = *(const uint4*)(vv + slab + (size_t)(cb * 64 + jp + 1) * 64 + sc);
            const bhalf* f0 = (const bhalf*)&v0;
            const bhalf* f1 = (const bhalf*)&v1;
#pragma unroll
            for (int t = 0; t < 8; ++t)
                *(unsigned int*)&Vs[sc + t][jp] =
                    (unsigned int)f0[t] | ((unsigned int)f1[t] << 16);
        }
        __syncthreads();

        // ---- S tile [16 rows x 64 cols] = Q_wave @ K_block ----
        v8f sacc[4];
#pragma unroll
        for (int nt = 0; nt < 4; ++nt) sacc[nt] = vzero;
#pragma unroll
        for (int ks = 0; ks < 2; ++ks) {        // K dim 64 -> 2 steps of 32
            const bhalf* ab = &Qs[wv * 16 + lrow][ks * 32 + lhi * 8];
            Frag a;
            a.q[0] = *(const uint4*)(ab);
            a.q[1] = *(const uint4*)(ab + 16);
#pragma unroll
            for (int nt = 0; nt < 4; ++nt) {
                const bhalf* bb = &Ks[nt * 16 + lrow][ks * 32 + lhi * 8];
                Frag bf;
                bf.q[0] = *(const uint4*)(bb);
                bf.q[1] = *(const uint4*)(bb + 16);
                sacc[nt] = __builtin_amdgcn_wmma_f32_16x16x32_bf16(
                    false, a.v, false, bf.v, (short)0, sacc[nt], false, false);
            }
        }

        // ---- online softmax update over these 64 columns ----
#pragma unroll
        for (int r = 0; r < 8; ++r) {
            float bm = fmaxf(fmaxf(sacc[0][r], sacc[1][r]),
                             fmaxf(sacc[2][r], sacc[3][r]));
            bm = fmaxf(bm, __shfl_xor(bm, 1, 32));
            bm = fmaxf(bm, __shfl_xor(bm, 2, 32));
            bm = fmaxf(bm, __shfl_xor(bm, 4, 32));
            bm = fmaxf(bm, __shfl_xor(bm, 8, 32));
            const float mnew  = fmaxf(rmax[r], bm);
            const float alpha = __expf(rmax[r] - mnew);
            rmax[r] = mnew;
            const int prow = r + 8 * lhi;       // matrix row of this r
            float ps = 0.f;
#pragma unroll
            for (int nt = 0; nt < 4; ++nt) {
                const float p = __expf(sacc[nt][r] - mnew);
                ps += p;
                Ps[wv][prow][nt * 16 + lrow] = f32_to_bf16(p);
                oacc[nt][r] *= alpha;
            }
            ps += __shfl_xor(ps, 1, 32);
            ps += __shfl_xor(ps, 2, 32);
            ps += __shfl_xor(ps, 4, 32);
            ps += __shfl_xor(ps, 8, 32);
            rsum[r] = rsum[r] * alpha + ps;
        }

        // ---- O += P[16x64] @ V_block[64x64] ----
#pragma unroll
        for (int ks = 0; ks < 2; ++ks) {
            const bhalf* ab = &Ps[wv][lrow][ks * 32 + lhi * 8];
            Frag a;
            a.q[0] = *(const uint4*)(ab);
            a.q[1] = *(const uint4*)(ab + 16);
#pragma unroll
            for (int nt = 0; nt < 4; ++nt) {
                const bhalf* bb = &Vs[nt * 16 + lrow][ks * 32 + lhi * 8];
                Frag bf;
                bf.q[0] = *(const uint4*)(bb);
                bf.q[1] = *(const uint4*)(bb + 16);
                oacc[nt] = __builtin_amdgcn_wmma_f32_16x16x32_bf16(
                    false, a.v, false, bf.v, (short)0, oacc[nt], false, false);
            }
        }
    }

    // ---- epilogue: O / rowsum -> ctx[b, w, h*64 + j] (head interleave) ----
#pragma unroll
    for (int r = 0; r < 8; ++r) {
        const float inv = 1.0f / rsum[r];
        const int w = m0 + wv * 16 + r + 8 * lhi;
        const size_t orow = ((size_t)b * 1024 + w) * 1024 + h * 64;
#pragma unroll
        for (int nt = 0; nt < 4; ++nt)
            ctx[orow + nt * 16 + lrow] = f32_to_bf16(oacc[nt][r] * inv);
    }
}

// ===========================================================================
extern "C" void kernel_launch(void* const* d_in, const int* in_sizes, int n_in,
                              void* d_out, int out_size, void* d_ws, size_t ws_size,
                              hipStream_t stream) {
    const float* k  = (const float*)d_in[0];
    const float* v  = (const float*)d_in[1];
    const float* q  = (const float*)d_in[2];
    const float* Wk = (const float*)d_in[3];
    const float* bk = (const float*)d_in[4];
    const float* Wv = (const float*)d_in[5];
    const float* bv = (const float*)d_in[6];
    const float* Wq = (const float*)d_in[7];
    const float* bq = (const float*)d_in[8];
    const float* Wo = (const float*)d_in[9];
    const float* bo = (const float*)d_in[10];
    float* out = (float*)d_out;

    const size_t NE = (size_t)8192 * 1024;
    bhalf* kkb = (bhalf*)d_ws;
    bhalf* vvb = kkb + NE;
    bhalf* qqb = vvb + NE;
    bhalf* ctx = qqb + NE;     // 4 * 16MB = 64MB of workspace

    dim3 blk(256);
    dim3 gg(64, 8);            // 8192/128 x 1024/128

    gemm_xWT<0, 0><<<gg, blk, 0, stream>>>((const void*)k, Wk, bk, (void*)kkb, 1.0f);
    gemm_xWT<0, 0><<<gg, blk, 0, stream>>>((const void*)v, Wv, bv, (void*)vvb, 1.0f);
    gemm_xWT<0, 0><<<gg, blk, 0, stream>>>((const void*)q, Wq, bq, (void*)qqb, 0.125f);  // 1/sqrt(64)

    attn_kernel<<<dim3(1024), blk, 0, stream>>>(kkb, vvb, qqb, ctx);

    gemm_xWT<1, 1><<<gg, blk, 0, stream>>>((const void*)ctx, Wo, bo, (void*)out, 1.0f);
}
